// FluxSingleAttention_33749853012421
// MI455X (gfx1250) — compile-verified, matrix operand order checked by reading, not run on hardware
//
#include <hip/hip_runtime.h>
#include <hip/hip_bf16.h>

#define SEQ   2048
#define DIM   3072
#define HEADS 24
#define HDIM  128

typedef __attribute__((ext_vector_type(16))) _Float16 v16h;
typedef __attribute__((ext_vector_type(8)))  _Float16 v8h;
typedef __attribute__((ext_vector_type(4)))  _Float16 v4h;
typedef __attribute__((ext_vector_type(8)))  float    v8f;

// ---------------------------------------------------------------------------
// helpers
// ---------------------------------------------------------------------------
static __device__ __forceinline__ v16h ldfrag(const _Float16* p0, const _Float16* p1) {
    v8h lo = *(const v8h*)p0;
    v8h hi = *(const v8h*)p1;
    v16h r;
#pragma unroll
    for (int i = 0; i < 8; ++i) { r[i] = lo[i]; r[i + 8] = hi[i]; }
    return r;
}

static __device__ __forceinline__ v8f wmma32(v16h a, v16h b, v8f c) {
    // D = A(16x32 f16) * B(32x16 f16) + C(16x16 f32)
    return __builtin_amdgcn_wmma_f32_16x16x32_f16(false, a, false, b, (short)0, c,
                                                  false, false);
}

// ---------------------------------------------------------------------------
// 1) cast x (f32) -> f16, 4 elems/thread
// ---------------------------------------------------------------------------
__global__ void cast_to_f16(const float* __restrict__ in, _Float16* __restrict__ out) {
    int i = (blockIdx.x * blockDim.x + threadIdx.x) * 4;
    float4 v = *(const float4*)(in + i);
    v4h o = { (_Float16)v.x, (_Float16)v.y, (_Float16)v.z, (_Float16)v.w };
    *(v4h*)(out + i) = o;
}

// ---------------------------------------------------------------------------
// 2) W [K,N] f32 -> Wt [N,K] f16 (tiled transpose, coalesced both sides)
// ---------------------------------------------------------------------------
__global__ void transpose_cast_w(const float* __restrict__ W, _Float16* __restrict__ Wt) {
    __shared__ float tile[32][33];
    int tx = threadIdx.x, ty = threadIdx.y;
    int n = blockIdx.x * 32 + tx;
    int k = blockIdx.y * 32 + ty;
    tile[ty][tx] = W[k * DIM + n];
    __syncthreads();
    int n2 = blockIdx.x * 32 + ty;
    int k2 = blockIdx.y * 32 + tx;
    Wt[n2 * DIM + k2] = (_Float16)tile[tx][ty];
}

// ---------------------------------------------------------------------------
// 3) C[S,3072] = X[S,3072] @ W + bias   (A: X f16, B: Wt f16 = W^T row-major)
//    One wave -> 32x64 output tile (2x4 sub-tiles), K stepped by 32.
// ---------------------------------------------------------------------------
__global__ __launch_bounds__(128) void gemm_bias(const _Float16* __restrict__ X,
                                                 const _Float16* __restrict__ Wt,
                                                 const float* __restrict__ bias,
                                                 float* __restrict__ C) {
    const int lane = threadIdx.x & 31;
    const int wave = blockIdx.x * 4 + (threadIdx.x >> 5);
    const int NT = DIM / 64;           // 48 col-tiles
    const int tm = wave / NT;          // 0..63  (2048/32)
    const int tn = wave % NT;
    const int ml = lane & 15, hi = lane >> 4;
    const int row0 = tm * 32, col0 = tn * 64;

    const _Float16* a0p = X + (size_t)(row0 + ml) * DIM;
    const _Float16* a1p = a0p + 16 * DIM;
    const _Float16* bp0 = Wt + (size_t)(col0 + 0 * 16 + ml) * DIM;
    const _Float16* bp1 = Wt + (size_t)(col0 + 1 * 16 + ml) * DIM;
    const _Float16* bp2 = Wt + (size_t)(col0 + 2 * 16 + ml) * DIM;
    const _Float16* bp3 = Wt + (size_t)(col0 + 3 * 16 + ml) * DIM;
    const _Float16* bp[4] = { bp0, bp1, bp2, bp3 };

    v8f acc[2][4] = {};
    for (int k = 0; k < DIM; k += 32) {
        const int o0 = k + hi * 8;
        const int o1 = k + 16 + hi * 8;
        v16h a0 = ldfrag(a0p + o0, a0p + o1);
        v16h a1 = ldfrag(a1p + o0, a1p + o1);
#pragma unroll
        for (int j = 0; j < 4; ++j) {
            v16h b = ldfrag(bp[j] + o0, bp[j] + o1);
            acc[0][j] = wmma32(a0, b, acc[0][j]);
            acc[1][j] = wmma32(a1, b, acc[1][j]);
        }
    }
#pragma unroll
    for (int i = 0; i < 2; ++i)
#pragma unroll
        for (int j = 0; j < 4; ++j) {
            int col = col0 + j * 16 + ml;
            float bv = bias[col];
#pragma unroll
            for (int r = 0; r < 8; ++r)
                C[(size_t)(row0 + i * 16 + 8 * hi + r) * DIM + col] = acc[i][j][r] + bv;
        }
}

// ---------------------------------------------------------------------------
// 4a) per-(s,head) RMSNorm + gamma + RoPE, f32 -> f16. One wave per row of 128.
// ---------------------------------------------------------------------------
__global__ void qk_norm_rope(const float* __restrict__ in, const float* __restrict__ gamma,
                             const float* __restrict__ cosb, const float* __restrict__ sinb,
                             _Float16* __restrict__ out) {
    const int lane = threadIdx.x & 31;
    const int wave = blockIdx.x * (blockDim.x >> 5) + (threadIdx.x >> 5);
    const int s = wave / HEADS;
    const int h = wave % HEADS;
    const float* row = in + (size_t)s * DIM + h * HDIM;
    const int d0 = lane * 4;
    float x0 = row[d0], x1 = row[d0 + 1], x2 = row[d0 + 2], x3 = row[d0 + 3];
    float ss = x0 * x0 + x1 * x1 + x2 * x2 + x3 * x3;
#pragma unroll
    for (int st = 1; st < 32; st <<= 1) ss += __shfl_xor(ss, st, 32);
    const float inv = rsqrtf(ss * (1.0f / 128.0f) + 1e-6f);
    x0 *= inv * gamma[d0];
    x1 *= inv * gamma[d0 + 1];
    x2 *= inv * gamma[d0 + 2];
    x3 *= inv * gamma[d0 + 3];
    const float* cp = cosb + (size_t)s * HDIM + d0;
    const float* sp = sinb + (size_t)s * HDIM + d0;
    float c0 = cp[0], c1 = cp[1], c2 = cp[2], c3 = cp[3];
    float s0 = sp[0], s1 = sp[1], s2 = sp[2], s3 = sp[3];
    float y0 = x0 * c0 - x1 * s0;
    float y1 = x1 * c1 + x0 * s1;
    float y2 = x2 * c2 - x3 * s2;
    float y3 = x3 * c3 + x2 * s3;
    _Float16* op = out + (size_t)s * DIM + h * HDIM + d0;
    v4h o = { (_Float16)y0, (_Float16)y1, (_Float16)y2, (_Float16)y3 };
    *(v4h*)op = o;
}

// ---------------------------------------------------------------------------
// 4b) V [S, H*D] f32 -> Vt [H][D][S] f16
// ---------------------------------------------------------------------------
__global__ void v_transpose_f16(const float* __restrict__ Vf, _Float16* __restrict__ Vt) {
    int idx = blockIdx.x * blockDim.x + threadIdx.x;   // h*128*2048 + d*2048 + s
    int s = idx & (SEQ - 1);
    int d = (idx >> 11) & (HDIM - 1);
    int h = idx >> 18;
    Vt[idx] = (_Float16)Vf[(size_t)s * DIM + h * HDIM + d];
}

// ---------------------------------------------------------------------------
// 5) Flash attention: one wave per (head, 16-row q-tile), streaming 32 keys/iter.
// ---------------------------------------------------------------------------
__global__ __launch_bounds__(128) void flash_attn(const _Float16* __restrict__ Qh,
                                                  const _Float16* __restrict__ Kh,
                                                  const _Float16* __restrict__ Vt,
                                                  float* __restrict__ Out) {
    __shared__ _Float16 pbuf[4][16][32];   // per-wave P tile (16 q x 32 keys)
    const int lane = threadIdx.x & 31;
    const int wid  = threadIdx.x >> 5;
    const int task = blockIdx.x * 4 + wid;     // HEADS * (SEQ/16) tasks
    const int nQT  = SEQ / 16;
    const int h  = task / nQT;
    const int qt = task % nQT;
    const int ml = lane & 15, hi = lane >> 4;
    const float scale = 0.08838834764831845f;  // 1/sqrt(128)

    // preload Q A-fragments (16 rows x 128 d, 4 chunks of K=32)
    const _Float16* qrow = Qh + (size_t)(qt * 16 + ml) * DIM + h * HDIM;
    v16h qf[4];
#pragma unroll
    for (int c = 0; c < 4; ++c)
        qf[c] = ldfrag(qrow + c * 32 + hi * 8, qrow + c * 32 + 16 + hi * 8);

    v8f acc[8] = {};
    float m[8], l[8];
#pragma unroll
    for (int r = 0; r < 8; ++r) { m[r] = -3.0e38f; l[r] = 0.0f; }

    const _Float16* vbase = Vt + (size_t)h * HDIM * SEQ;

    for (int kb = 0; kb < SEQ; kb += 32) {
        // ---- scores for 32 keys: two 16x16 WMMA tiles, Kdim=d=128 ----
        v8f sc0 = {}, sc1 = {};
        const _Float16* kr0 = Kh + (size_t)(kb + ml) * DIM + h * HDIM;
        const _Float16* kr1 = kr0 + 16 * DIM;
#pragma unroll
        for (int c = 0; c < 4; ++c) {
            v16h b0 = ldfrag(kr0 + c * 32 + hi * 8, kr0 + c * 32 + 16 + hi * 8);
            sc0 = wmma32(qf[c], b0, sc0);
            v16h b1 = ldfrag(kr1 + c * 32 + hi * 8, kr1 + c * 32 + 16 + hi * 8);
            sc1 = wmma32(qf[c], b1, sc1);
        }
        // ---- online softmax (rows = r + 8*hi, cols spread over 16 lanes) ----
        float f0[8], f1[8], rmx[8], p0[8], p1[8], rs[8], al[8];
#pragma unroll
        for (int r = 0; r < 8; ++r) {
            f0[r] = sc0[r] * scale;
            f1[r] = sc1[r] * scale;
            rmx[r] = fmaxf(f0[r], f1[r]);
        }
#pragma unroll
        for (int st = 1; st < 16; st <<= 1)
#pragma unroll
            for (int r = 0; r < 8; ++r)
                rmx[r] = fmaxf(rmx[r], __shfl_xor(rmx[r], st, 32));
#pragma unroll
        for (int r = 0; r < 8; ++r) {
            float mn = fmaxf(m[r], rmx[r]);
            al[r] = __expf(m[r] - mn);
            p0[r] = __expf(f0[r] - mn);
            p1[r] = __expf(f1[r] - mn);
            rs[r] = p0[r] + p1[r];
            m[r]  = mn;
        }
#pragma unroll
        for (int st = 1; st < 16; st <<= 1)
#pragma unroll
            for (int r = 0; r < 8; ++r)
                rs[r] += __shfl_xor(rs[r], st, 32);
#pragma unroll
        for (int r = 0; r < 8; ++r) l[r] = l[r] * al[r] + rs[r];

        // ---- re-pack P into A-fragment layout through LDS (same-wave DS: in-order) ----
#pragma unroll
        for (int r = 0; r < 8; ++r) {
            pbuf[wid][r + 8 * hi][ml]      = (_Float16)p0[r];
            pbuf[wid][r + 8 * hi][16 + ml] = (_Float16)p1[r];
        }
        const _Float16* prow = &pbuf[wid][ml][0];
        v16h pa = ldfrag(prow + hi * 8, prow + 16 + hi * 8);

        // ---- rescale O and accumulate P @ V over 8 d-tiles ----
#pragma unroll
        for (int j = 0; j < 8; ++j) {
#pragma unroll
            for (int r = 0; r < 8; ++r) acc[j][r] *= al[r];
            const _Float16* vr = vbase + (size_t)(j * 16 + ml) * SEQ + kb;
            v16h bv = ldfrag(vr + hi * 8, vr + 16 + hi * 8);
            acc[j] = wmma32(pa, bv, acc[j]);
        }
    }
    // ---- normalize and store ----
#pragma unroll
    for (int j = 0; j < 8; ++j)
#pragma unroll
        for (int r = 0; r < 8; ++r)
            Out[(size_t)(qt * 16 + r + 8 * hi) * DIM + h * HDIM + j * 16 + ml] =
                acc[j][r] / l[r];
}

// ---------------------------------------------------------------------------
// host
// ---------------------------------------------------------------------------
extern "C" void kernel_launch(void* const* d_in, const int* in_sizes, int n_in,
                              void* d_out, int out_size, void* d_ws, size_t ws_size,
                              hipStream_t stream) {
    (void)in_sizes; (void)n_in; (void)out_size; (void)ws_size;
    const float* x    = (const float*)d_in[0];
    const float* cosb = (const float*)d_in[1];
    const float* sinb = (const float*)d_in[2];
    const float* Wq   = (const float*)d_in[3];
    const float* bq   = (const float*)d_in[4];
    const float* Wk   = (const float*)d_in[5];
    const float* bk   = (const float*)d_in[6];
    const float* Wv   = (const float*)d_in[7];
    const float* bv   = (const float*)d_in[8];
    const float* gq   = (const float*)d_in[9];
    const float* gk   = (const float*)d_in[10];

    char* ws = (char*)d_ws;
    size_t off = 0;
    auto alloc = [&](size_t bytes) {
        void* p = ws + off;
        off += (bytes + 255) & ~(size_t)255;
        return p;
    };
    const size_t SD = (size_t)SEQ * DIM;
    _Float16* xh  = (_Float16*)alloc(SD * 2);
    _Float16* wtq = (_Float16*)alloc((size_t)DIM * DIM * 2);
    _Float16* wtk = (_Float16*)alloc((size_t)DIM * DIM * 2);
    _Float16* wtv = (_Float16*)alloc((size_t)DIM * DIM * 2);
    float*    Qf  = (float*)alloc(SD * 4);
    float*    Kf  = (float*)alloc(SD * 4);
    float*    Vf  = (float*)alloc(SD * 4);
    _Float16* qh  = (_Float16*)alloc(SD * 2);
    _Float16* kh  = (_Float16*)alloc(SD * 2);
    _Float16* vt  = (_Float16*)alloc(SD * 2);

    // 1) x -> f16
    cast_to_f16<<<(SD / 4) / 256, 256, 0, stream>>>(x, xh);

    // 2) weight transpose+cast
    dim3 tg(DIM / 32, DIM / 32), tb(32, 32);
    transpose_cast_w<<<tg, tb, 0, stream>>>(Wq, wtq);
    transpose_cast_w<<<tg, tb, 0, stream>>>(Wk, wtk);
    transpose_cast_w<<<tg, tb, 0, stream>>>(Wv, wtv);

    // 3) QKV GEMMs: (2048/32)*(3072/64) = 3072 waves, 4 waves/block
    const int gemmBlocks = (SEQ / 32) * (DIM / 64) / 4;   // 768
    gemm_bias<<<gemmBlocks, 128, 0, stream>>>(xh, wtq, bq, Qf);
    gemm_bias<<<gemmBlocks, 128, 0, stream>>>(xh, wtk, bk, Kf);
    gemm_bias<<<gemmBlocks, 128, 0, stream>>>(xh, wtv, bv, Vf);

    // 4) norm + rope (S*HEADS waves, 8 waves/block) and V transpose
    const int nrBlocks = SEQ * HEADS / 8;                 // 6144
    qk_norm_rope<<<nrBlocks, 256, 0, stream>>>(Qf, gq, cosb, sinb, qh);
    qk_norm_rope<<<nrBlocks, 256, 0, stream>>>(Kf, gk, cosb, sinb, kh);
    v_transpose_f16<<<SD / 256, 256, 0, stream>>>(Vf, vt);

    // 5) flash attention: HEADS*(SEQ/16) = 3072 waves, 4 waves/block
    const int faBlocks = HEADS * (SEQ / 16) / 4;          // 768
    flash_attn<<<faBlocks, 128, 0, stream>>>(qh, kh, vt, (float*)d_out);
}
